// MultiFreqAttention_60129542719
// MI455X (gfx1250) — matrix operand order repaired and checked by reading, not compiled, not run on hardware
//
#include <hip/hip_runtime.h>
#include <cstdint>

// MultiFreqAttention for MI455X (gfx1250, wave32, WMMA).
// GEMM A tiles are DMA'd to LDS by the Tensor Data Mover (tensor_load_to_lds,
// TENSORcnt) using the D# pad feature to produce the padded LDS layout.
// Attention K tiles use global_load_async_to_lds_b128 (ASYNCcnt).
// B tiles are stored in LDS pre-swizzled into WMMA fragment order so fragment
// loads are contiguous ds_load_b128.

typedef __attribute__((ext_vector_type(16))) _Float16 v16h;
typedef __attribute__((ext_vector_type(8)))  float    v8f;
typedef __attribute__((ext_vector_type(4)))  uint32_t su4;
typedef __attribute__((ext_vector_type(8)))  uint32_t su8;

#define DEV __device__ __forceinline__

DEV v8f wmma16(v16h a, v16h b, v8f c) {
  // (neg_a, A, neg_b, B, c_mod, C, reuse_a, reuse_b)
  return __builtin_amdgcn_wmma_f32_16x16x32_f16(false, a, false, b, (short)0, c,
                                                false, false);
}

// CDNA5 async global->LDS copy, 16B per lane. INST_OFFSET applies to both the
// global and LDS addresses (ISA 08_async_tensor §4.4).
DEV void async_ld_b128(uint32_t lds_off, const void* gaddr) {
  asm volatile("global_load_async_to_lds_b128 %0, %1, off"
               :: "v"(lds_off), "v"(gaddr) : "memory");
}
DEV void async_ld_b128_o16(uint32_t lds_off, const void* gaddr) {
  asm volatile("global_load_async_to_lds_b128 %0, %1, off offset:16"
               :: "v"(lds_off), "v"(gaddr) : "memory");
}
DEV void async_ld_b128_o32(uint32_t lds_off, const void* gaddr) {
  asm volatile("global_load_async_to_lds_b128 %0, %1, off offset:32"
               :: "v"(lds_off), "v"(gaddr) : "memory");
}
DEV void async_ld_b128_o48(uint32_t lds_off, const void* gaddr) {
  asm volatile("global_load_async_to_lds_b128 %0, %1, off offset:48"
               :: "v"(lds_off), "v"(gaddr) : "memory");
}
DEV void wait_async0() { asm volatile("s_wait_asynccnt 0x0" ::: "memory"); }

// TDM: load a 2D tile (tile_dim0 x tile_dim1 elements of 2B) from a row-major
// tensor with row stride `stride` (elements) into LDS at byte offset lds_addr,
// inserting `padDw` DWORDs of LDS padding after every 2^(padIntLog2+1) DWORDs.
// D# bitfields per CDNA5 ISA (2D: groups 2/3 absent).
DEV void tdm_load_2d(uint32_t lds_addr, const void* gaddr, uint32_t tile0,
                     uint32_t tile1, uint32_t dim0, uint32_t dim1,
                     uint32_t stride, uint32_t padIntCode, uint32_t padAmtCode) {
  uint64_t va = (uint64_t)(uintptr_t)gaddr;
  su4 g0;
  g0[0] = 1u;                                   // count=1 (valid descriptor)
  g0[1] = lds_addr;                             // lds_addr [63:32]
  g0[2] = (uint32_t)va;                         // global_addr [95:64]
  g0[3] = ((uint32_t)(va >> 32) & 0x01FFFFFFu)  // global_addr [120:96]
          | (2u << 30);                         // type=2 ("image")
  su8 g1;
  g1[0] = (1u << 16)                            // data_size=1 -> 2 bytes
          | (1u << 20)                          // pad_enable
          | (padIntCode << 22)                  // pad_interval
          | (padAmtCode << 25);                 // pad_amount
  g1[1] = (dim0 & 0xFFFFu) << 16;               // tensor_dim0[15:0] @ bit48
  g1[2] = ((dim0 >> 16) & 0xFFFFu)              // tensor_dim0[31:16]
          | ((dim1 & 0xFFFFu) << 16);           // tensor_dim1[15:0]
  g1[3] = ((dim1 >> 16) & 0xFFFFu)              // tensor_dim1[31:16]
          | (tile0 << 16);                      // tile_dim0 @ bit112
  g1[4] = tile1;                                // tile_dim1 @ bit128
  g1[5] = stride;                               // tensor_dim0_stride[31:0]
  g1[6] = 0u;                                   // stride hi / dim1_stride lo
  g1[7] = 0u;                                   // dim1_stride hi
  asm volatile("tensor_load_to_lds %0, %1" :: "s"(g0), "s"(g1) : "memory");
}

// A fragment (16x32 f16): lane L -> row m0+(L&15); K [kb..kb+15]
// with kb = (L>>4)*16 (contiguous order, matching B).
DEV v16h load_a_frag(const _Float16* p, int ld, int m0, int k0, int lane) {
  const _Float16* q = p + (size_t)(m0 + (lane & 15)) * ld + k0 + ((lane >> 4) << 4);
  v16h a;
#pragma unroll
  for (int i = 0; i < 16; ++i) a[i] = q[i];
  return a;
}

// B fragment from swizzled storage: cell stride 24 halfs (48B, 16B-aligned),
// lane L reads 16 contiguous halfs -> two ds_load_b128.
DEV v16h load_b_sw(const _Float16* cell24, int lane) {
  const _Float16* q = cell24 + (size_t)lane * 24;
  v16h b;
#pragma unroll
  for (int j = 0; j < 16; ++j) b[j] = q[j];
  return b;
}

// ---------------------------------------------------------------------------
// Generic tiled WMMA GEMM: C[bz] = A[bz](MxK) * B[bz](KxN)  (+bias[n]) (*g+h)
// Block = 256 threads (8 waves), tile 128x128, K-step 32.
// Requires: M % 128 == 0, K % 32 == 0, N % 16 == 0 (N tile edge guarded).
// ---------------------------------------------------------------------------
__global__ __launch_bounds__(256) void gemm_wmma_kernel(
    const _Float16* __restrict__ A, const _Float16* __restrict__ B,
    const float* __restrict__ bias, const float* __restrict__ gh,
    float* __restrict__ Cf, _Float16* __restrict__ Ch,
    int M, int N, int K,
    long long bsA, long long bsB, long long bsC, int relu) {
  __shared__ alignas(16) _Float16 As[128][40];    // A tile, TDM-padded rows
  __shared__ alignas(16) _Float16 Bsw[8][32][24]; // fragment-swizzled B tile
  const int tid = threadIdx.x;
  const int lane = tid & 31;
  const int wid = tid >> 5;
  const int wr = wid & 3;        // wave row: 4 x 32 rows
  const int wc = wid >> 2;       // wave col: 2 x 64 cols
  const int nbase = blockIdx.x * 128;
  const int mbase = blockIdx.y * 128;
  const int bz = blockIdx.z;
  const _Float16* Ab = A + (long long)bz * bsA;
  const _Float16* Bb = B + (long long)bz * bsB;

  v8f acc[2][4];
#pragma unroll
  for (int i = 0; i < 2; ++i)
#pragma unroll
    for (int j = 0; j < 4; ++j) acc[i][j] = {};

  const int brow = tid >> 3, bcol = (tid & 7) << 4;   // 32 x 128 halfs
  const int gn = tid & 7;                              // B column group
  const uint32_t lds_a = (uint32_t)(size_t)&As[0][0];

  for (int kt = 0; kt < K; kt += 32) {
    // --- A tile: Tensor Data Mover DMA to LDS (TENSORcnt), wave 0 only.
    // 32x128-element tile, row stride K, LDS pad +4 DWORDs per 16 DWORDs
    // reproduces the As[128][40] layout. ---
    if (wid == 0) {
      const _Float16* srcA = Ab + (size_t)mbase * K + kt;
      tdm_load_2d(lds_a, srcA, /*tile0=*/32u, /*tile1=*/128u,
                  /*dim0=*/(uint32_t)(K - kt), /*dim1=*/(uint32_t)M,
                  /*stride=*/(uint32_t)K, /*padInt=*/3u, /*padAmt=*/3u);
    }
    // --- B tile: load row segment, scatter into fragment order ---
    {
      union { uint4 q[2]; _Float16 h[16]; } u;
      if (nbase + bcol < N) {
        const _Float16* src = Bb + (size_t)(kt + brow) * N + nbase + bcol;
        const uint4* s4 = reinterpret_cast<const uint4*>(src);
        u.q[0] = s4[0]; u.q[1] = s4[1];
        if (kt + 32 < K) __builtin_prefetch(src + (size_t)32 * N, 0, 1);
      } else {
        u.q[0] = make_uint4(0u, 0u, 0u, 0u);
        u.q[1] = make_uint4(0u, 0u, 0u, 0u);
      }
      const int Lb = brow & 16, jj = brow & 15;
#pragma unroll
      for (int t = 0; t < 16; ++t) Bsw[gn][Lb + t][jj] = u.h[t];
    }
    if (wid == 0) __builtin_amdgcn_s_wait_tensorcnt(0);
    __syncthreads();

    v16h bf[4];
#pragma unroll
    for (int ni = 0; ni < 4; ++ni)
      bf[ni] = load_b_sw(&Bsw[wc * 4 + ni][0][0], lane);
#pragma unroll
    for (int mi = 0; mi < 2; ++mi) {
      v16h af = load_a_frag(&As[0][0], 40, wr * 32 + mi * 16, 0, lane);
#pragma unroll
      for (int ni = 0; ni < 4; ++ni) acc[mi][ni] = wmma16(af, bf[ni], acc[mi][ni]);
    }
    __syncthreads();
  }

  float g = 1.f, hh = 0.f;
  if (gh) { g = gh[2 * bz]; hh = gh[2 * bz + 1]; }
#pragma unroll
  for (int mi = 0; mi < 2; ++mi) {
#pragma unroll
    for (int ni = 0; ni < 4; ++ni) {
#pragma unroll
      for (int r = 0; r < 8; ++r) {
        int m = mbase + wr * 32 + mi * 16 + r + ((lane >> 4) << 3);
        int n = nbase + wc * 64 + ni * 16 + (lane & 15);
        if (n < N) {
          float x = acc[mi][ni][r];
          if (bias) x += bias[n];
          x = x * g + hh;
          if (relu) x = fmaxf(x, 0.f);
          long long o = (long long)bz * bsC + (size_t)m * N + n;
          if (Cf) Cf[o] = x;
          if (Ch) Ch[o] = (_Float16)x;
        }
      }
    }
  }
}

// ---------------------------------------------------------------------------
// Window attention: 1 wave per (window, head). N=64 tokens, hd=32.
// K tile staged to LDS via async global->LDS DMA (ASYNCcnt); V staged into
// fragment-swizzled LDS. S = (q*scale) k^T + bias -> softmax -> O = P v,
// all matrix work in WMMA. Accumulates per-batch sum/sumsq of O (src stats).
// ---------------------------------------------------------------------------
__global__ __launch_bounds__(32) void attn_kernel(
    const _Float16* __restrict__ qkv,    // (4096*64, 288) [q(96)|k(96)|v(96)]
    const float* __restrict__ bias64,    // (64,64,3)
    _Float16* __restrict__ awh,          // (4096*64, 96)
    float* __restrict__ stats) {
  __shared__ alignas(16) float    Sb[64][65];
  __shared__ alignas(16) _Float16 Pb[64][66];
  __shared__ alignas(16) _Float16 Vsw[2][2][32][24]; // [k0/32][dgrp][lane][j]
  __shared__ alignas(16) _Float16 Ksh[64][32];       // K tile (row, d)
  const int w = blockIdx.x / 3;
  const int h = blockIdx.x % 3;
  const int lane = threadIdx.x;
  const _Float16* base = qkv + (size_t)w * 64 * 288;
  const float qscale = 0.17677669529663689f;  // 32^-0.5

  // ---- stage K tile via async DMA: 2 rows (2 x 64B) per lane ----
  {
    const uint32_t ksh0 = (uint32_t)(size_t)&Ksh[0][0];
#pragma unroll
    for (int rr = 0; rr < 2; ++rr) {
      int key = lane * 2 + rr;
      uint32_t ldsk = ksh0 + (uint32_t)key * 64u;
      const _Float16* gk = base + (size_t)key * 288 + 96 + h * 32;
      async_ld_b128(ldsk, gk);
      async_ld_b128_o16(ldsk, gk);
      async_ld_b128_o32(ldsk, gk);
      async_ld_b128_o48(ldsk, gk);
    }
  }

  // ---- stage V into fragment-swizzled LDS (2 rows per lane) ----
#pragma unroll
  for (int rr = 0; rr < 2; ++rr) {
    int key = lane * 2 + rr;
    union { uint4 q[2]; _Float16 h2[32]; } u;
    const uint4* vp =
        reinterpret_cast<const uint4*>(base + (size_t)key * 288 + 192 + h * 32);
    u.q[0] = vp[0]; u.q[1] = vp[1];
    int ks = key >> 5;
    int Lr = ((key >> 4) & 1) << 4;
    int jj = key & 15;
#pragma unroll
    for (int d = 0; d < 32; ++d) Vsw[ks][d >> 4][Lr + (d & 15)][jj] = u.h2[d];
  }
  wait_async0();
  __syncthreads();

  // ---- S = q k^T + bias ----
#pragma unroll
  for (int mt = 0; mt < 4; ++mt) {
    v16h a;
    {
      const _Float16* qp =
          base + (size_t)(mt * 16 + (lane & 15)) * 288 + h * 32 + ((lane >> 4) << 4);
#pragma unroll
      for (int i = 0; i < 16; ++i) {
        a[i]     = (_Float16)((float)qp[i] * qscale);
      }
    }
#pragma unroll
    for (int nt = 0; nt < 4; ++nt) {
      // B[d][key] = k[key][d] -> contiguous 16 halfs of the key row in LDS
      v16h bf;
      const _Float16* kp = &Ksh[0][0] +
          (size_t)(nt * 16 + (lane & 15)) * 32 + ((lane >> 4) << 4);
#pragma unroll
      for (int j = 0; j < 16; ++j) bf[j] = kp[j];
      v8f c = {};
      c = wmma16(a, bf, c);
#pragma unroll
      for (int r = 0; r < 8; ++r) {
        int mm = mt * 16 + r + ((lane >> 4) << 3);
        int nn = nt * 16 + (lane & 15);
        Sb[mm][nn] = c[r] + bias64[(mm * 64 + nn) * 3 + h];
      }
    }
  }
  __syncthreads();

  // ---- softmax, 2 rows per lane ----
#pragma unroll
  for (int rr = 0; rr < 2; ++rr) {
    int row = lane * 2 + rr;
    float mx = -3.4e38f;
    for (int j = 0; j < 64; ++j) mx = fmaxf(mx, Sb[row][j]);
    float sum = 0.f;
    for (int j = 0; j < 64; ++j) { float e = __expf(Sb[row][j] - mx); Sb[row][j] = e; sum += e; }
    float inv = 1.f / sum;
    for (int j = 0; j < 64; ++j) Pb[row][j] = (_Float16)(Sb[row][j] * inv);
  }
  __syncthreads();

  // ---- O = P v ----
  float s1 = 0.f, s2 = 0.f;
#pragma unroll
  for (int mt = 0; mt < 4; ++mt) {
#pragma unroll
    for (int nt = 0; nt < 2; ++nt) {
      v8f c = {};
#pragma unroll
      for (int ks = 0; ks < 2; ++ks) {
        v16h a = load_a_frag(&Pb[0][0], 66, mt * 16, ks * 32, lane);
        v16h bf = load_b_sw(&Vsw[ks][nt][0][0], lane);
        c = wmma16(a, bf, c);
      }
#pragma unroll
      for (int r = 0; r < 8; ++r) {
        int mm = mt * 16 + r + ((lane >> 4) << 3);
        int nn = nt * 16 + (lane & 15);
        float v = c[r];
        awh[(size_t)(w * 64 + mm) * 96 + h * 32 + nn] = (_Float16)v;
        s1 += v; s2 += v * v;
      }
    }
  }
  atomicAdd(&stats[w >> 10], s1);
  atomicAdd(&stats[4 + (w >> 10)], s2);
}

// ---------------------------------------------------------------------------
// Scalar helper kernels
// ---------------------------------------------------------------------------
DEV int refl256(int p) { p = p < 0 ? -p : p; return p > 255 ? 510 - p : p; }
DEV float slog1p(float d) {
  return d > 0.f ? log1pf(d) : (d < 0.f ? -log1pf(-d) : 0.f);
}

__global__ void zero_kernel(float* s, int n) {
  int i = blockIdx.x * blockDim.x + threadIdx.x;
  if (i < n) s[i] = 0.f;
}

// relative-position MLP bias -> (64,64,3)
__global__ void prep_bias_kernel(const float* __restrict__ w1, const float* __restrict__ b1,
                                 const float* __restrict__ w2, const float* __restrict__ b2,
                                 float* __restrict__ bias64) {
  int p = blockIdx.x * blockDim.x + threadIdx.x;
  if (p >= 4096) return;
  int nq = p >> 6, nk = p & 63;
  float r0 = slog1p((float)((nq >> 3) - (nk >> 3)));
  float r1 = slog1p((float)((nq & 7) - (nk & 7)));
  float a0 = 0.f, a1 = 0.f, a2 = 0.f;
  for (int u = 0; u < 256; ++u) {
    float hh = fmaxf(r0 * w1[u * 2] + r1 * w1[u * 2 + 1] + b1[u], 0.f);
    a0 += hh * w2[u];
    a1 += hh * w2[256 + u];
    a2 += hh * w2[512 + u];
  }
  bias64[p * 3 + 0] = a0 + b2[0];
  bias64[p * 3 + 1] = a1 + b2[1];
  bias64[p * 3 + 2] = a2 + b2[2];
}

// fused qkv weight (96x288, B-layout) + bias, torch cat order [qk(192), v(96)]
__global__ void prep_wqkv_kernel(const float* __restrict__ qkw, const float* __restrict__ qkb,
                                 const float* __restrict__ vw, const float* __restrict__ vb,
                                 _Float16* __restrict__ wq, float* __restrict__ qb) {
  int idx = blockIdx.x * blockDim.x + threadIdx.x;
  if (idx < 96 * 288) {
    int k = idx / 288, n = idx % 288;
    float v = (n < 192) ? qkw[n * 96 + k] : vw[(n - 192) * 96 + k];
    wq[idx] = (_Float16)v;
  }
  if (idx < 288) qb[idx] = (idx < 192) ? qkb[idx] : vb[idx - 192];
}

// pe_w1 (256,6144) -> f16 transposed (6144,256)
__global__ void prep_b1h_kernel(const float* __restrict__ w, _Float16* __restrict__ o) {
  size_t idx = (size_t)blockIdx.x * blockDim.x + threadIdx.x;
  if (idx < (size_t)6144 * 256) {
    int k = (int)(idx / 256), n = (int)(idx % 256);
    o[idx] = (_Float16)w[(size_t)n * 6144 + k];
  }
}

// X (B,C,H,W) -> window-pixel-major f16 (262144, 96)
__global__ void prep_xwin_kernel(const float* __restrict__ X, _Float16* __restrict__ xw) {
  for (size_t idx = (size_t)blockIdx.x * blockDim.x + threadIdx.x; idx < 25165824ULL;
       idx += (size_t)gridDim.x * blockDim.x) {
    int c = (int)(idx % 96);
    size_t wrow = idx / 96;
    int n = (int)(wrow % 64);
    int w = (int)(wrow / 64);
    int b = w >> 10, pw = w & 1023;
    int i = pw >> 5, j = pw & 31;
    int y = (i << 3) + (n >> 3), x = (j << 3) + (n & 7);
    xw[idx] = (_Float16)X[(size_t)(b * 96 + c) * 65536 + y * 256 + x];
  }
}

// depthwise 5x5, reflect pad, + bias, optional relu
__global__ void dwconv5_kernel(const float* __restrict__ in, const float* __restrict__ w,
                               const float* __restrict__ bias, float* __restrict__ out,
                               int relu) {
  for (size_t idx = (size_t)blockIdx.x * blockDim.x + threadIdx.x; idx < 25165824ULL;
       idx += (size_t)gridDim.x * blockDim.x) {
    int x = (int)(idx & 255), y = (int)((idx >> 8) & 255);
    int bc = (int)(idx >> 16);
    int c = bc % 96;
    const float* inp = in + ((size_t)bc << 16);
    const float* wc = w + c * 25;
    float acc = bias[c];
#pragma unroll
    for (int dy = 0; dy < 5; ++dy) {
      int ry = refl256(y + dy - 2);
#pragma unroll
      for (int dx = 0; dx < 5; ++dx) {
        int rx = refl256(x + dx - 2);
        acc += inp[(ry << 8) + rx] * wc[dy * 5 + dx];
      }
    }
    if (relu) acc = fmaxf(acc, 0.f);
    out[idx] = acc;
  }
}

// hid (4096,256) f16 -> emb (4096,32) f32 l2-normalized + cur stats atomics
__global__ __launch_bounds__(64) void emb_kernel(const _Float16* __restrict__ hid,
                                                 const float* __restrict__ w2,
                                                 const float* __restrict__ b2,
                                                 float* __restrict__ emb,
                                                 float* __restrict__ stats) {
  __shared__ float hrow[256];
  __shared__ float vals[32];
  int p = blockIdx.x, t = threadIdx.x;
  const _Float16* hp = hid + (size_t)p * 256;
  for (int i = t; i < 256; i += 64) hrow[i] = (float)hp[i];
  __syncthreads();
  float d = 0.f;
  if (t < 30) {
    const float* wp = w2 + t * 256;
    for (int u = 0; u < 256; ++u) d += hrow[u] * wp[u];
    d += b2[t];
  }
  if (t < 32) vals[t] = (t < 30) ? d : 0.f;
  __syncthreads();
  if (t < 32) {
    float ss = 0.f;
    for (int i = 0; i < 30; ++i) ss += vals[i] * vals[i];
    float nv = vals[t] / (sqrtf(ss) + 1e-8f);
    emb[(size_t)p * 32 + t] = (t < 30) ? nv : 0.f;
    if (t < 30) {
      int b = p >> 10;
      atomicAdd(&stats[8 + b], nv);
      atomicAdd(&stats[12 + b], nv * nv);
    }
  }
}

// per-batch AdaIN scale/shift: g = src_std/cur_std, h = src_mean - cur_mean*g
__global__ void finalize_kernel(float* __restrict__ s) {
  int t = threadIdx.x;
  if (t >= 4) return;
  const float Ns = 1024.f * 6144.f, Nc = 1024.f * 30.f;
  float sm = s[t] / Ns;
  float sstd = sqrtf(fmaxf(s[4 + t] / Ns - sm * sm, 0.f) + 1e-8f);
  float cm = s[8 + t] / Nc;
  float cstd = sqrtf(fmaxf(s[12 + t] / Nc - cm * cm, 0.f) + 1e-8f);
  float g = sstd / cstd;
  s[16 + 2 * t] = g;
  s[17 + 2 * t] = sm - cm * g;
}

// row of exp-similarity attention, l1-normalized, f16 (one block per row)
__global__ __launch_bounds__(256) void pg_kernel(const float* __restrict__ emb,
                                                 const float* __restrict__ alpha_p,
                                                 const float* __restrict__ beta_p,
                                                 _Float16* __restrict__ pg) {
  __shared__ float em[32];
  __shared__ float vbuf[1024];
  __shared__ float red[256];
  int pm = blockIdx.x;       // global patch row 0..4095
  int b = pm >> 10;
  int t = threadIdx.x;
  if (t < 32) em[t] = emb[(size_t)pm * 32 + t];
  __syncthreads();
  float alpha = 30.f * alpha_p[0];
  float beta = 20.f * beta_p[0];
  float psum = 0.f;
  for (int j = t; j < 1024; j += 256) {
    const float* ej = emb + (size_t)(b * 1024 + j) * 32;
    float dt = 0.f;
#pragma unroll
    for (int d0 = 0; d0 < 30; ++d0) dt += em[d0] * ej[d0];
    float v = __expf(alpha * dt + beta);
    vbuf[j] = v;
    psum += v;
  }
  red[t] = psum;
  __syncthreads();
  for (int s = 128; s > 0; s >>= 1) { if (t < s) red[t] += red[t + s]; __syncthreads(); }
  float inv = 1.f / (red[0] + 1e-8f);
  __syncthreads();
  _Float16* outp = pg + (size_t)pm * 1024;
  for (int j = t; j < 1024; j += 256) outp[j] = (_Float16)(vbuf[j] * inv);
}

// feats = dwconv5(hf1, w2) + b2 + window_reverse(rw)   (writes d_out)
__global__ void fuse_kernel(const float* __restrict__ hf1, const float* __restrict__ w2,
                            const float* __restrict__ b2, const float* __restrict__ rw,
                            float* __restrict__ out) {
  for (size_t idx = (size_t)blockIdx.x * blockDim.x + threadIdx.x; idx < 25165824ULL;
       idx += (size_t)gridDim.x * blockDim.x) {
    int x = (int)(idx & 255), y = (int)((idx >> 8) & 255);
    int bc = (int)(idx >> 16);
    int c = bc % 96, b = bc / 96;
    const float* inp = hf1 + ((size_t)bc << 16);
    const float* wc = w2 + c * 25;
    float acc = b2[c];
#pragma unroll
    for (int dy = 0; dy < 5; ++dy) {
      int ry = refl256(y + dy - 2);
#pragma unroll
      for (int dx = 0; dx < 5; ++dx) {
        int rx = refl256(x + dx - 2);
        acc += inp[(ry << 8) + rx] * wc[dy * 5 + dx];
      }
    }
    int patch = (y >> 3) * 32 + (x >> 3);
    int feat = (((y & 7) << 3) + (x & 7)) * 96 + c;
    acc += rw[(size_t)(b * 1024 + patch) * 6144 + feat];
    out[idx] = acc;
  }
}

// fusion conv1: 96->48, 3x3 stride2 pad1, relu
__global__ void fconv1_kernel(const float* __restrict__ in, const float* __restrict__ w1,
                              float* __restrict__ a1) {
  for (size_t idx = (size_t)blockIdx.x * blockDim.x + threadIdx.x; idx < 3145728ULL;
       idx += (size_t)gridDim.x * blockDim.x) {
    int x = (int)(idx & 127), y = (int)((idx >> 7) & 127);
    int bf = (int)(idx >> 14);
    int f = bf % 48, b = bf / 48;
    float acc = 0.f;
    const float* inb = in + (size_t)b * 96 * 65536;
    for (int c = 0; c < 96; ++c) {
      const float* inc = inb + (size_t)c * 65536;
      const float* wp = w1 + (f * 96 + c) * 9;
#pragma unroll
      for (int ky = 0; ky < 3; ++ky) {
        int iy = 2 * y + ky - 1;
        if ((unsigned)iy >= 256u) continue;
#pragma unroll
        for (int kx = 0; kx < 3; ++kx) {
          int ix = 2 * x + kx - 1;
          if ((unsigned)ix >= 256u) continue;
          acc += inc[(iy << 8) + ix] * wp[ky * 3 + kx];
        }
      }
    }
    a1[idx] = fmaxf(acc, 0.f);
  }
}

// fusion conv2 (48->96, stride2) fused with spatial mean -> a2 (4,96)
__global__ __launch_bounds__(256) void fconv2_mean_kernel(const float* __restrict__ a1,
                                                          const float* __restrict__ w2,
                                                          float* __restrict__ a2) {
  __shared__ float red[256];
  int boc = blockIdx.x;
  int oc = boc % 96, b = boc / 96;
  int tid = threadIdx.x;
  float acc = 0.f;
  const float* inb = a1 + (size_t)b * 48 * 16384;
  for (int sp = tid; sp < 4096; sp += 256) {
    int y = sp >> 6, x = sp & 63;
    float v = 0.f;
    for (int g = 0; g < 48; ++g) {
      const float* ing = inb + (size_t)g * 16384;
      const float* wp = w2 + (oc * 48 + g) * 9;
#pragma unroll
      for (int ky = 0; ky < 3; ++ky) {
        int iy = 2 * y + ky - 1;
        if ((unsigned)iy >= 128u) continue;
#pragma unroll
        for (int kx = 0; kx < 3; ++kx) {
          int ix = 2 * x + kx - 1;
          if ((unsigned)ix >= 128u) continue;
          v += ing[(iy << 7) + ix] * wp[ky * 3 + kx];
        }
      }
    }
    acc += v;
  }
  red[tid] = acc;
  __syncthreads();
  for (int s = 128; s > 0; s >>= 1) { if (tid < s) red[tid] += red[tid + s]; __syncthreads(); }
  if (tid == 0) a2[boc] = red[0] * (1.f / 4096.f);
}

// channel softmax gate
__global__ __launch_bounds__(128) void gate_kernel(const float* __restrict__ a2,
                                                   float* __restrict__ gate) {
  __shared__ float buf[128];
  int b = blockIdx.x, t = threadIdx.x;
  float v = (t < 96) ? a2[b * 96 + t] : -3.4e38f;
  buf[t] = v;
  __syncthreads();
  for (int s = 64; s > 0; s >>= 1) { if (t < s) buf[t] = fmaxf(buf[t], buf[t + s]); __syncthreads(); }
  float mx = buf[0];
  __syncthreads();
  float e = (t < 96) ? __expf(v - mx) : 0.f;
  buf[t] = e;
  __syncthreads();
  for (int s = 64; s > 0; s >>= 1) { if (t < s) buf[t] += buf[t + s]; __syncthreads(); }
  float sum = buf[0];
  if (t < 96) gate[b * 96 + t] = e / sum;
}

__global__ void scale_kernel(float* __restrict__ out, const float* __restrict__ gate) {
  for (size_t idx = (size_t)blockIdx.x * blockDim.x + threadIdx.x; idx < 25165824ULL;
       idx += (size_t)gridDim.x * blockDim.x) {
    int bc = (int)(idx >> 16);  // == b*96 + c
    out[idx] *= gate[bc];
  }
}

// ---------------------------------------------------------------------------
extern "C" void kernel_launch(void* const* d_in, const int* in_sizes, int n_in,
                              void* d_out, int out_size, void* d_ws, size_t ws_size,
                              hipStream_t stream) {
  (void)in_sizes; (void)n_in; (void)out_size; (void)ws_size;
  const float* X       = (const float*)d_in[0];
  const float* hf_w1   = (const float*)d_in[1];
  const float* hf_b1   = (const float*)d_in[2];
  const float* hf_w2   = (const float*)d_in[3];
  const float* hf_b2   = (const float*)d_in[4];
  const float* lfv_w   = (const float*)d_in[5];
  const float* lfv_b   = (const float*)d_in[6];
  const float* lfqk_w  = (const float*)d_in[7];
  const float* lfqk_b  = (const float*)d_in[8];
  const float* meta_w1 = (const float*)d_in[9];
  const float* meta_b1 = (const float*)d_in[10];
  const float* meta_w2 = (const float*)d_in[11];
  const float* meta_b2 = (const float*)d_in[12];
  const float* pe_w1   = (const float*)d_in[13];
  const float* pe_b1   = (const float*)d_in[14];
  const float* pe_w2   = (const float*)d_in[15];
  const float* pe_b2   = (const float*)d_in[16];
  const float* att_a   = (const float*)d_in[17];
  const float* att_b   = (const float*)d_in[18];
  const float* fus_w1  = (const float*)d_in[19];
  const float* fus_w2  = (const float*)d_in[20];
  float* out = (float*)d_out;

  char* base = (char*)d_ws;
  size_t off = 0;
  auto alloc = [&](size_t bytes) -> void* {
    off = (off + 255) & ~(size_t)255;
    void* r = base + off;
    off += bytes;
    return r;
  };
  float*    bias64 = (float*)alloc((size_t)64 * 64 * 3 * 4);
  _Float16* wqkvh  = (_Float16*)alloc((size_t)96 * 288 * 2);
  float*    qkvb   = (float*)alloc(288 * 4);
  _Float16* b1h    = (_Float16*)alloc((size_t)6144 * 256 * 2);
  float*    stats  = (float*)alloc(64 * 4);
  _Float16* hid    = (_Float16*)alloc((size_t)4096 * 256 * 2);
  float*    emb    = (float*)alloc((size_t)4096 * 32 * 4);
  _Float16* pg     = (_Float16*)alloc((size_t)4096 * 1024 * 2);
  float*    a1     = (float*)alloc((size_t)4 * 48 * 128 * 128 * 4);
  float*    a2     = (float*)alloc((size_t)4 * 96 * 4);
  float*    gate   = (float*)alloc((size_t)4 * 96 * 4);
  _Float16* xwin   = (_Float16*)alloc(25165824ULL * 2);
  _Float16* qkvh   = (_Float16*)alloc((size_t)262144 * 288 * 2);
  _Float16* awh    = (_Float16*)alloc(25165824ULL * 2);
  float*    hf1    = (float*)alloc(25165824ULL * 4);
  float*    rw     = (float*)alloc(25165824ULL * 4);

  // prep
  zero_kernel<<<1, 64, 0, stream>>>(stats, 64);
  prep_bias_kernel<<<16, 256, 0, stream>>>(meta_w1, meta_b1, meta_w2, meta_b2, bias64);
  prep_wqkv_kernel<<<108, 256, 0, stream>>>(lfqk_w, lfqk_b, lfv_w, lfv_b, wqkvh, qkvb);
  prep_b1h_kernel<<<6144, 256, 0, stream>>>(pe_w1, b1h);
  prep_xwin_kernel<<<4096, 256, 0, stream>>>(X, xwin);

  // high-frequency branch (conv1; conv2 is fused into fuse_kernel)
  dwconv5_kernel<<<8192, 256, 0, stream>>>(X, hf_w1, hf_b1, hf1, 1);

  // QKV projection: (262144 x 96) * (96 x 288) -> f16
  gemm_wmma_kernel<<<dim3(3, 2048, 1), 256, 0, stream>>>(
      xwin, wqkvh, qkvb, nullptr, nullptr, qkvh, 262144, 288, 96, 0, 0, 0, 0);

  // window attention (also accumulates src stats)
  attn_kernel<<<12288, 32, 0, stream>>>(qkvh, bias64, awh, stats);

  // patch embed GEMM1: (4096 x 6144) * (6144 x 256), bias+relu -> f16
  gemm_wmma_kernel<<<dim3(2, 32, 1), 256, 0, stream>>>(
      awh, b1h, pe_b1, nullptr, nullptr, hid, 4096, 256, 6144, 0, 0, 0, 1);

  // embed GEMM2 + l2 norm + cur stats
  emb_kernel<<<4096, 64, 0, stream>>>(hid, pe_w2, pe_b2, emb, stats);
  finalize_kernel<<<1, 32, 0, stream>>>(stats);

  // exp-similarity attention matrix, l1-normalized, f16
  pg_kernel<<<4096, 256, 0, stream>>>(emb, att_a, att_b, pg);

  // rw = P_g @ aw with fused AdaIN epilogue (per-batch g,h in stats[16..23])
  gemm_wmma_kernel<<<dim3(48, 8, 4), 256, 0, stream>>>(
      pg, awh, nullptr, stats + 16, rw, nullptr, 1024, 6144, 1024,
      1024LL * 1024, 1024LL * 6144, 1024LL * 6144, 0);

  // feats_sum = hf + window_reverse(rw) -> d_out; then SK fusion gate
  fuse_kernel<<<8192, 256, 0, stream>>>(hf1, hf_w2, hf_b2, rw, out);
  fconv1_kernel<<<4096, 256, 0, stream>>>(out, fus_w1, a1);
  fconv2_mean_kernel<<<384, 256, 0, stream>>>(a1, fus_w2, a2);
  gate_kernel<<<4, 128, 0, stream>>>(a2, gate);
  scale_kernel<<<8192, 256, 0, stream>>>(out, gate);
}
